// LSTMFloodClassifier_45999099740405
// MI455X (gfx1250) — matrix operand order repaired and checked by reading, not compiled, not run on hardware
//
#include <hip/hip_runtime.h>
#include <hip/hip_bf16.h>
#include <math.h>

// ---------------------------------------------------------------------------
// BiLSTM(2 layers, H=128) + MLP head for MI455X (gfx1250, wave32, WMMA).
//  * recurrent weights LDS-resident (bf16, fragment-packed), cell state in regs
//  * per-step GEMM via v_wmma_f32_16x16x32_bf16, f32 accumulate
//  * all WMMA fragments loaded as 2x b128 (pre-packed B tiles, contiguous A)
//  * TDM (tensor_load_to_lds, 6-arg form) stages the 128KB recurrent weights
//  * layer-1 backward direction = single step (only h2[-1] is consumed)
//  * activation phase: per-thread hh-pair ownership, loop-invariant biases in
//    registers, per-step x scalars staged once into LDS, packed 32-bit h stores
// Workspace requirement: ~358 MB (dominated by the (T,B,256) bf16 h1 buffer).
// ---------------------------------------------------------------------------

typedef __attribute__((ext_vector_type(16))) __bf16 v16bf;
typedef __attribute__((ext_vector_type(8)))  float  v8f;
typedef __attribute__((ext_vector_type(4)))  unsigned int u32x4;
typedef __attribute__((ext_vector_type(8)))  int i32x8;
typedef __attribute__((ext_vector_type(4)))  int i32x4;

#define Hh    128
#define Gg    512            // 4*H
#define Tlen  168
#define Bsz   4096
#define BT    64             // batch rows per block
#define NTHR  256            // 8 wave32

#if defined(__gfx1250__) && __has_builtin(__builtin_amdgcn_tensor_load_to_lds) && \
    __has_builtin(__builtin_amdgcn_s_wait_tensorcnt)
#define HAVE_TDM 1
#else
#define HAVE_TDM 0
#endif

// ---- fast activations (guarded hardware transcendentals) ------------------
static __device__ __forceinline__ float tanh_(float x) {
#if __has_builtin(__builtin_amdgcn_tanhf)
  return __builtin_amdgcn_tanhf(x);
#else
  return tanhf(x);
#endif
}
static __device__ __forceinline__ float sigm_(float x) {
#if __has_builtin(__builtin_amdgcn_exp2f)
  return 1.0f / (1.0f + __builtin_amdgcn_exp2f(-1.44269504f * x));
#else
  return 1.0f / (1.0f + __expf(-x));
#endif
}

// ---- wave32 WMMA fragment helpers (layouts per CDNA5 ISA 7.12.2) ----------
union frag16 { uint4 u[2]; v16bf v; };
union frag8f { uint4 u[2]; v8f v; };
union bfpair { __bf16 h[2]; unsigned u; };

// A: 16x32 bf16 (MxK) from row-major src; each lane = two contiguous 16B runs.
static __device__ __forceinline__ v16bf load_A_bf16(const __bf16* src, int ld) {
  const int lane = threadIdx.x & 31;
  const int half = lane >> 4, m = lane & 15;
  const __bf16* p = src + m * ld + half * 8;
  frag16 f;
  f.u[0] = *(const uint4*)(p);
  f.u[1] = *(const uint4*)(p + 16);
  return f.v;
}
// B: fragment-packed tiles: base[((tile*32 + lane)*16 + j] (see pack kernel).
static __device__ __forceinline__ v16bf load_B_packed(const __bf16* base, int tile) {
  const int lane = threadIdx.x & 31;
  const uint4* p = (const uint4*)(base + ((size_t)tile * 32 + lane) * 16);
  frag16 f;
  f.u[0] = p[0];
  f.u[1] = p[1];
  return f.v;
}
// C/D 16x16 f32, stored tile-packed: base[(tile*32 + lane)*8 + r].
static __device__ __forceinline__ void store_C_packed(float* base, int tile, v8f c) {
  const int lane = threadIdx.x & 31;
  frag8f f;
  f.v = c;
  uint4* p = (uint4*)(base + ((size_t)tile * 32 + lane) * 8);
  p[0] = f.u[0];
  p[1] = f.u[1];
}
// read element (b, n) of the packed 64x512 gates buffer (tile = (b>>4)*32 + (n>>4))
static __device__ __forceinline__ float read_gate(const float* base, int b, int n) {
  const int mi = b >> 4, m = b & 15;
  const int ni = n >> 4, nn = n & 15;
  const int lane = ((m >> 3) << 4) + nn;
  const int r = m & 7;
  return base[(((mi << 5) + ni) * 32 + lane) * 8 + r];
}

#define WMMA_BF16(A, B, C) \
  __builtin_amdgcn_wmma_f32_16x16x32_bf16(false, (A), false, (B), (short)0, (C), false, false)

// ---- TDM: 1-D bulk copy global -> LDS via Tensor Data Mover ---------------
#if HAVE_TDM
static __device__ __forceinline__ void tdm_load_lds(unsigned lds_off, const void* gsrc,
                                                    unsigned bytes) {
  const unsigned long long ga = (unsigned long long)(size_t)gsrc;
  const unsigned units = bytes >> 3;  // data_size = 8 bytes
  u32x4 g0;
  i32x8 g1;
  i32x4 g2 = {0, 0, 0, 0}, g3 = {0, 0, 0, 0};
  i32x8 g4 = {0, 0, 0, 0, 0, 0, 0, 0};
  g0[0] = 1u;                                               // count=1 (valid, user)
  g0[1] = lds_off;                                          // D#.lds_addr
  g0[2] = (unsigned)ga;                                     // global_addr[31:0]
  g0[3] = (unsigned)((ga >> 32) & 0x1FFFFFFu) | (2u << 30); // [56:32] | type=2
  g1[0] = (int)(3u << 16);                                  // data_size=8B, mask=0
  g1[1] = (int)((units & 0xFFFFu) << 16);                   // tensor_dim0[15:0]
  g1[2] = (int)(((units >> 16) & 0xFFFFu) | (1u << 16));    // dim0[31:16], dim1=1
  g1[3] = (int)((units & 0xFFFFu) << 16);                   // tile_dim0
  g1[4] = 1;                                                // tile_dim1=1
  g1[5] = (int)units;                                       // dim0_stride[31:0]
  g1[6] = 0;
  g1[7] = 0;
  __builtin_amdgcn_tensor_load_to_lds(g0, g1, g2, g3, g4, 0);
}
#endif

// ---------------------------------------------------------------------------
// prep: fp32 (512 x K) row-major -> bf16 B-fragment-packed W^T (K x 512).
__global__ void pack_b_frag_bf16(__bf16* __restrict__ dst, const float* __restrict__ src,
                                 int K) {
  const int i = blockIdx.x * blockDim.x + threadIdx.x;
  if (i >= 512 * K) return;
  const int j    = i & 15;
  const int l    = (i >> 4) & 31;
  const int tile = i >> 9;
  const int ni   = tile & 31;
  const int kt   = tile >> 5;
  const int n    = ni * 16 + (l & 15);
  const int k    = kt * 32 + (j & 8) * 2 + (l >> 4) * 8 + (j & 7);
  dst[i] = (__bf16)src[n * K + k];
}

// ---------------------------------------------------------------------------
// Layer 0 BiLSTM (input dim 1). grid = (B/BT, 2[dir]); block = 256 threads.
// dyn smem: sW 128KB + sH 16KB + sG 128KB + sXv 256B = 278784 B. sW at LDS 0.
__global__ void lstm_l0(const float* __restrict__ x,                   // (B,T,1)
                        const float* __restrict__ w0f, const float* __restrict__ w0b,
                        const __bf16* __restrict__ WTf, const __bf16* __restrict__ WTb,
                        const float* __restrict__ bihf, const float* __restrict__ bhhf,
                        const float* __restrict__ bihb, const float* __restrict__ bhhb,
                        __bf16* __restrict__ H1)                       // (T,B,256)
{
  extern __shared__ char smem[];
  __bf16* sW  = (__bf16*)smem;                             // packed, 128*512 bf16
  __bf16* sH  = (__bf16*)(smem + 131072);                  // 64*128 bf16 row-major
  float*  sG  = (float*)(smem + 131072 + 16384);           // packed gates
  float*  sXv = (float*)(smem + 131072 + 16384 + 131072);  // 64 x-scalars

  const int dir = blockIdx.y;
  const int b0  = blockIdx.x * BT;
  const int tid = threadIdx.x;
  const __bf16* WT  = dir ? WTb : WTf;
  const float*  w0  = dir ? w0b : w0f;
  const float*  bih = dir ? bihb : bihf;
  const float*  bhh = dir ? bhhb : bhhf;

#if HAVE_TDM
  if (tid < 32) tdm_load_lds(0u, WT, Hh * Gg * 2);   // wave 0 issues the DMA
#else
  {
    const uint4* s = (const uint4*)WT;
    uint4* d = (uint4*)sW;
    for (int i = tid; i < (Hh * Gg * 2) / 16; i += NTHR) d[i] = s[i];
  }
#endif
  for (int i = tid; i < BT * Hh; i += NTHR) sH[i] = (__bf16)0.0f;

  // per-thread ownership: hh pair (2*(tid&63), +1), batch rows b = 4e + (tid>>6)
  const int hh0 = (tid & 63) * 2;
  const int q   = tid >> 6;
  // loop-invariant gate biases and input taps (registers)
  const float bi0 = bih[hh0]       + bhh[hh0],       bi1 = bih[hh0 + 1]       + bhh[hh0 + 1];
  const float bf0 = bih[128 + hh0] + bhh[128 + hh0], bf1 = bih[128 + hh0 + 1] + bhh[128 + hh0 + 1];
  const float bg0 = bih[256 + hh0] + bhh[256 + hh0], bg1 = bih[256 + hh0 + 1] + bhh[256 + hh0 + 1];
  const float bo0 = bih[384 + hh0] + bhh[384 + hh0], bo1 = bih[384 + hh0 + 1] + bhh[384 + hh0 + 1];
  const float wi0 = w0[hh0],       wi1 = w0[hh0 + 1];
  const float wf0 = w0[128 + hh0], wf1 = w0[128 + hh0 + 1];
  const float wg0 = w0[256 + hh0], wg1 = w0[256 + hh0 + 1];
  const float wo0 = w0[384 + hh0], wo1 = w0[384 + hh0 + 1];

  float2 c2[16];
#pragma unroll
  for (int e = 0; e < 16; ++e) c2[e] = make_float2(0.0f, 0.0f);
#if HAVE_TDM
  __builtin_amdgcn_s_wait_tensorcnt(0);
#endif
  __syncthreads();

  const int w   = tid >> 5;
  const int mi  = w & 3;   // M tile owned by this wave
  const int nio = w >> 2;  // N interleave (0/1)

  for (int t = 0; t < Tlen; ++t) {
    const int tt = dir ? (Tlen - 1 - t) : t;

    // stage this step's 64 x scalars (one per batch row) into LDS
    if (tid < BT) sXv[tid] = x[(size_t)(b0 + tid) * Tlen + tt];

    // gates(64x512) = h(64x128) @ WhhT(128x512)
    v16bf a0 = load_A_bf16(sH + (mi * 16) * Hh + 0,  Hh);
    v16bf a1 = load_A_bf16(sH + (mi * 16) * Hh + 32, Hh);
    v16bf a2 = load_A_bf16(sH + (mi * 16) * Hh + 64, Hh);
    v16bf a3 = load_A_bf16(sH + (mi * 16) * Hh + 96, Hh);
#pragma unroll 1
    for (int s = 0; s < 16; ++s) {
      const int ni = nio + 2 * s;
      v8f acc = {};
      v16bf bb;
      bb = load_B_packed(sW, 0 * 32 + ni); acc = WMMA_BF16(a0, bb, acc);
      bb = load_B_packed(sW, 1 * 32 + ni); acc = WMMA_BF16(a1, bb, acc);
      bb = load_B_packed(sW, 2 * 32 + ni); acc = WMMA_BF16(a2, bb, acc);
      bb = load_B_packed(sW, 3 * 32 + ni); acc = WMMA_BF16(a3, bb, acc);
      store_C_packed(sG, mi * 32 + ni, acc);
    }
    __syncthreads();

    // activations + state update
#pragma unroll
    for (int e = 0; e < 16; ++e) {
      const int b = 4 * e + q;
      const float xs = sXv[b];
      const float gi0 = read_gate(sG, b, hh0)           + bi0 + xs * wi0;
      const float gi1 = read_gate(sG, b, hh0 + 1)       + bi1 + xs * wi1;
      const float gf0 = read_gate(sG, b, 128 + hh0)     + bf0 + xs * wf0;
      const float gf1 = read_gate(sG, b, 128 + hh0 + 1) + bf1 + xs * wf1;
      const float gg0 = read_gate(sG, b, 256 + hh0)     + bg0 + xs * wg0;
      const float gg1 = read_gate(sG, b, 256 + hh0 + 1) + bg1 + xs * wg1;
      const float go0 = read_gate(sG, b, 384 + hh0)     + bo0 + xs * wo0;
      const float go1 = read_gate(sG, b, 384 + hh0 + 1) + bo1 + xs * wo1;
      const float cc0 = sigm_(gf0) * c2[e].x + sigm_(gi0) * tanh_(gg0);
      const float cc1 = sigm_(gf1) * c2[e].y + sigm_(gi1) * tanh_(gg1);
      c2[e] = make_float2(cc0, cc1);
      bfpair p;
      p.h[0] = (__bf16)(sigm_(go0) * tanh_(cc0));
      p.h[1] = (__bf16)(sigm_(go1) * tanh_(cc1));
      *(unsigned*)(sH + b * Hh + hh0) = p.u;
      *(unsigned*)(H1 + (size_t)tt * Bsz * 256 + (size_t)(b0 + b) * 256 + dir * Hh + hh0) = p.u;
    }
    __syncthreads();
  }
}

// ---------------------------------------------------------------------------
// Layer 1 BiLSTM (input dim 256). dir 0: full scan; dir 1: single step.
// dyn smem: sW 128KB + sX 32KB + sH 16KB + sG 128KB = 311296 B. sW at offset 0.
__global__ void lstm_l1(const __bf16* __restrict__ H1,                 // (T,B,256)
                        const __bf16* __restrict__ WihTf, const __bf16* __restrict__ WihTb,
                        const __bf16* __restrict__ WhhTf, const __bf16* __restrict__ WhhTb,
                        const float* __restrict__ bihf, const float* __restrict__ bhhf,
                        const float* __restrict__ bihb, const float* __restrict__ bhhb,
                        float* __restrict__ h2last)                    // (B,256) f32
{
  extern __shared__ char smem[];
  __bf16* sW = (__bf16*)smem;                            // packed, 128*512 bf16
  __bf16* sX = (__bf16*)(smem + 131072);                 // 64*256 bf16 row-major
  __bf16* sH = (__bf16*)(smem + 131072 + 32768);         // 64*128 bf16 row-major
  float*  sG = (float*)(smem + 131072 + 32768 + 16384);  // packed gates

  const int dir = blockIdx.y;
  const int b0  = blockIdx.x * BT;
  const int tid = threadIdx.x;
  const __bf16* WihT = dir ? WihTb : WihTf;  // packed (8x32 tiles), L2 resident
  const __bf16* WhhT = dir ? WhhTb : WhhTf;
  const float*  bih  = dir ? bihb : bihf;
  const float*  bhh  = dir ? bhhb : bhhf;

#if HAVE_TDM
  if (tid < 32) tdm_load_lds(0u, WhhT, Hh * Gg * 2);
#else
  {
    const uint4* s = (const uint4*)WhhT;
    uint4* d = (uint4*)sW;
    for (int i = tid; i < (Hh * Gg * 2) / 16; i += NTHR) d[i] = s[i];
  }
#endif
  for (int i = tid; i < BT * Hh; i += NTHR) sH[i] = (__bf16)0.0f;

  const int hh0 = (tid & 63) * 2;
  const int q   = tid >> 6;
  const float bi0 = bih[hh0]       + bhh[hh0],       bi1 = bih[hh0 + 1]       + bhh[hh0 + 1];
  const float bf0 = bih[128 + hh0] + bhh[128 + hh0], bf1 = bih[128 + hh0 + 1] + bhh[128 + hh0 + 1];
  const float bg0 = bih[256 + hh0] + bhh[256 + hh0], bg1 = bih[256 + hh0 + 1] + bhh[256 + hh0 + 1];
  const float bo0 = bih[384 + hh0] + bhh[384 + hh0], bo1 = bih[384 + hh0 + 1] + bhh[384 + hh0 + 1];

  float2 c2[16];
#pragma unroll
  for (int e = 0; e < 16; ++e) c2[e] = make_float2(0.0f, 0.0f);
#if HAVE_TDM
  __builtin_amdgcn_s_wait_tensorcnt(0);
#endif
  __syncthreads();

  const int w   = tid >> 5;
  const int mi  = w & 3;
  const int nio = w >> 2;
  const int steps = dir ? 1 : Tlen;

  for (int t = 0; t < steps; ++t) {
    const int tt = dir ? (Tlen - 1) : t;

    {  // stage x_t tile (64x256 bf16) into LDS, coalesced b128
      const uint4* s = (const uint4*)(H1 + (size_t)tt * Bsz * 256 + (size_t)b0 * 256);
      uint4* d = (uint4*)sX;
      for (int i = tid; i < (BT * 256 * 2) / 16; i += NTHR) d[i] = s[i];
    }
    if (t + 1 < steps) {  // prefetch next timestep's input tile
      const char* nxt = (const char*)(H1 + (size_t)(tt + 1) * Bsz * 256 + (size_t)b0 * 256);
      __builtin_prefetch(nxt + tid * 128, 0, 1);
    }
    __syncthreads();

    // gates(64x512) = x_t(64x256)@WihT(256x512) + h(64x128)@WhhT(128x512)
    v16bf ax[8], ah[4];
#pragma unroll
    for (int kt = 0; kt < 8; ++kt) ax[kt] = load_A_bf16(sX + (mi * 16) * 256 + kt * 32, 256);
#pragma unroll
    for (int kt = 0; kt < 4; ++kt) ah[kt] = load_A_bf16(sH + (mi * 16) * Hh + kt * 32, Hh);
#pragma unroll 1
    for (int s = 0; s < 16; ++s) {
      const int ni = nio + 2 * s;
      v8f acc = {};
#pragma unroll
      for (int kt = 0; kt < 8; ++kt) {
        v16bf bb = load_B_packed(WihT, kt * 32 + ni);
        acc = WMMA_BF16(ax[kt], bb, acc);
      }
#pragma unroll
      for (int kt = 0; kt < 4; ++kt) {
        v16bf bb = load_B_packed(sW, kt * 32 + ni);
        acc = WMMA_BF16(ah[kt], bb, acc);
      }
      store_C_packed(sG, mi * 32 + ni, acc);
    }
    __syncthreads();

#pragma unroll
    for (int e = 0; e < 16; ++e) {
      const int b = 4 * e + q;
      const float gi0 = read_gate(sG, b, hh0)           + bi0;
      const float gi1 = read_gate(sG, b, hh0 + 1)       + bi1;
      const float gf0 = read_gate(sG, b, 128 + hh0)     + bf0;
      const float gf1 = read_gate(sG, b, 128 + hh0 + 1) + bf1;
      const float gg0 = read_gate(sG, b, 256 + hh0)     + bg0;
      const float gg1 = read_gate(sG, b, 256 + hh0 + 1) + bg1;
      const float go0 = read_gate(sG, b, 384 + hh0)     + bo0;
      const float go1 = read_gate(sG, b, 384 + hh0 + 1) + bo1;
      const float cc0 = sigm_(gf0) * c2[e].x + sigm_(gi0) * tanh_(gg0);
      const float cc1 = sigm_(gf1) * c2[e].y + sigm_(gi1) * tanh_(gg1);
      c2[e] = make_float2(cc0, cc1);
      const float hv0 = sigm_(go0) * tanh_(cc0);
      const float hv1 = sigm_(go1) * tanh_(cc1);
      bfpair p;
      p.h[0] = (__bf16)hv0;
      p.h[1] = (__bf16)hv1;
      *(unsigned*)(sH + b * Hh + hh0) = p.u;
      if (t == steps - 1)
        *(float2*)(h2last + (size_t)(b0 + b) * 256 + dir * Hh + hh0) = make_float2(hv0, hv1);
    }
    __syncthreads();
  }
}

// ---------------------------------------------------------------------------
// head: sigmoid(relu(last @ cw1.T + cb1) @ cw2.T + cb2). 1 block / batch row.
__global__ void classifier(const float* __restrict__ h2last,
                           const float* __restrict__ cw1, const float* __restrict__ cb1,
                           const float* __restrict__ cw2, const float* __restrict__ cb2,
                           float* __restrict__ out) {
  __shared__ float red[128];
  const int b = blockIdx.x;
  const int h = threadIdx.x;  // 128 threads
  const float* row = h2last + (size_t)b * 256;
  float z = cb1[h];
  const float* wrow = cw1 + (size_t)h * 256;
#pragma unroll 4
  for (int k = 0; k < 256; ++k) z = fmaf(row[k], wrow[k], z);
  z = fmaxf(z, 0.0f);
  red[h] = z * cw2[h];
  __syncthreads();
  for (int s = 64; s > 0; s >>= 1) {
    if (h < s) red[h] += red[h + s];
    __syncthreads();
  }
  if (h == 0) out[b] = 1.0f / (1.0f + __expf(-(red[0] + cb2[0])));
}

// ---------------------------------------------------------------------------
extern "C" void kernel_launch(void* const* d_in, const int* in_sizes, int n_in,
                              void* d_out, int out_size, void* d_ws, size_t ws_size,
                              hipStream_t stream) {
  const float* x        = (const float*)d_in[0];
  const float* w_ih_l0  = (const float*)d_in[1];   // (512,1)
  const float* w_hh_l0  = (const float*)d_in[2];   // (512,128)
  const float* b_ih_l0  = (const float*)d_in[3];
  const float* b_hh_l0  = (const float*)d_in[4];
  const float* w_ih_l0r = (const float*)d_in[5];
  const float* w_hh_l0r = (const float*)d_in[6];
  const float* b_ih_l0r = (const float*)d_in[7];
  const float* b_hh_l0r = (const float*)d_in[8];
  const float* w_ih_l1  = (const float*)d_in[9];   // (512,256)
  const float* w_hh_l1  = (const float*)d_in[10];
  const float* b_ih_l1  = (const float*)d_in[11];
  const float* b_hh_l1  = (const float*)d_in[12];
  const float* w_ih_l1r = (const float*)d_in[13];
  const float* w_hh_l1r = (const float*)d_in[14];
  const float* b_ih_l1r = (const float*)d_in[15];
  const float* b_hh_l1r = (const float*)d_in[16];
  const float* cw1      = (const float*)d_in[17];
  const float* cb1      = (const float*)d_in[18];
  const float* cw2      = (const float*)d_in[19];
  const float* cb2      = (const float*)d_in[20];

  char* ws = (char*)d_ws;
  size_t off = 0;
  __bf16* WhhT_l0f = (__bf16*)(ws + off); off += (size_t)Hh * Gg * 2;
  __bf16* WhhT_l0b = (__bf16*)(ws + off); off += (size_t)Hh * Gg * 2;
  __bf16* WhhT_l1f = (__bf16*)(ws + off); off += (size_t)Hh * Gg * 2;
  __bf16* WhhT_l1b = (__bf16*)(ws + off); off += (size_t)Hh * Gg * 2;
  __bf16* WihT_l1f = (__bf16*)(ws + off); off += (size_t)256 * Gg * 2;
  __bf16* WihT_l1b = (__bf16*)(ws + off); off += (size_t)256 * Gg * 2;
  float*  h2last   = (float*)(ws + off);  off += (size_t)Bsz * 256 * 4;
  __bf16* H1       = (__bf16*)(ws + off); off += (size_t)Tlen * Bsz * 256 * 2;

  // ---- weight prep: fp32 -> bf16, fragment-packed B tiles ----
  {
    const int n1 = Gg * Hh, n2 = Gg * 256;
    pack_b_frag_bf16<<<(n1 + 255) / 256, 256, 0, stream>>>(WhhT_l0f, w_hh_l0,  Hh);
    pack_b_frag_bf16<<<(n1 + 255) / 256, 256, 0, stream>>>(WhhT_l0b, w_hh_l0r, Hh);
    pack_b_frag_bf16<<<(n1 + 255) / 256, 256, 0, stream>>>(WhhT_l1f, w_hh_l1,  Hh);
    pack_b_frag_bf16<<<(n1 + 255) / 256, 256, 0, stream>>>(WhhT_l1b, w_hh_l1r, Hh);
    pack_b_frag_bf16<<<(n2 + 255) / 256, 256, 0, stream>>>(WihT_l1f, w_ih_l1,  256);
    pack_b_frag_bf16<<<(n2 + 255) / 256, 256, 0, stream>>>(WihT_l1b, w_ih_l1r, 256);
  }

  // ---- layer 0 (both directions concurrent via grid.y) ----
  {
    dim3 g(Bsz / BT, 2);
    lstm_l0<<<g, NTHR, 278784, stream>>>(x, w_ih_l0, w_ih_l0r, WhhT_l0f, WhhT_l0b,
                                         b_ih_l0, b_hh_l0, b_ih_l0r, b_hh_l0r, H1);
  }
  // ---- layer 1 (fwd full scan; bwd one step) ----
  {
    dim3 g(Bsz / BT, 2);
    lstm_l1<<<g, NTHR, 311296, stream>>>(H1, WihT_l1f, WihT_l1b, WhhT_l1f, WhhT_l1b,
                                         b_ih_l1, b_hh_l1, b_ih_l1r, b_hh_l1r, h2last);
  }
  // ---- classifier head ----
  classifier<<<Bsz, 128, 0, stream>>>(h2last, cw1, cb1, cw2, cb2, (float*)d_out);
}
